// WeightAdaptiveGallinear_24404004176741
// MI455X (gfx1250) — matrix operand order repaired and checked by reading, not compiled, not run on hardware
//
#include <hip/hip_runtime.h>
#include <hip/hip_bf16.h>
#include <math.h>

// ---------------- problem constants ----------------
#define IN_F   16
#define OUT_F  64
#define LAT    128
#define HID    1024
#define KK     6        // NH*NE
#define MM     1088     // (IN_F+1)*OUT_F
#define CS     6540     // (MM+2)*KK
#define CSP    6592     // CS padded to multiple of 64
#define BATCH  8192
#define XW     145      // IN_F + LAT + 1

typedef __bf16 bf16;
typedef __attribute__((ext_vector_type(8)))  __bf16 v8bf;
typedef __attribute__((ext_vector_type(16))) __bf16 v16bf;
typedef __attribute__((ext_vector_type(8)))  float  v8f;
typedef __attribute__((ext_vector_type(4)))  int    v4i;

// ---------------- async copy primitives (CDNA5 ASYNCcnt path) ----------------
__device__ __forceinline__ void async_copy16(const bf16* gp, bf16* lp) {
#if __has_builtin(__builtin_amdgcn_global_load_async_to_lds_b128)
    __builtin_amdgcn_global_load_async_to_lds_b128(
        (__attribute__((address_space(1))) v4i*)(uintptr_t)gp,
        (__attribute__((address_space(3))) v4i*)(void*)lp, 0, 0);
#else
    unsigned l = (unsigned)(uintptr_t)(__attribute__((address_space(3))) void*)(void*)lp;
    unsigned long long g = (unsigned long long)(uintptr_t)gp;
    asm volatile("global_load_async_to_lds_b128 %0, %1, off"
                 :: "v"(l), "v"(g) : "memory");
#endif
}

__device__ __forceinline__ void async_wait0() {
#if __has_builtin(__builtin_amdgcn_s_wait_asynccnt)
    __builtin_amdgcn_s_wait_asynccnt(0);
#else
    asm volatile("s_wait_asynccnt 0x0" ::: "memory");
#endif
}

// ---------------- WMMA fragment loaders (wave32, 16x16x32 bf16) ----------------
__device__ __forceinline__ v16bf load_frag_a_ptr(const bf16* base) {
    v8bf lo = *(const v8bf*)(base);
    v8bf hi = *(const v8bf*)(base + 16);
    return __builtin_shufflevector(lo, hi, 0,1,2,3,4,5,6,7,8,9,10,11,12,13,14,15);
}

// A matrix 16(M)x32(K), row-major. Lanes 0-15: M=lane, K 0..7/16..23; lanes 16-31: M=lane-16, K 8..15/24..31.
__device__ __forceinline__ v16bf load_frag_a(const bf16* A, int lda, int m0, int k0) {
    int lane = threadIdx.x & 31;
    int m    = m0 + (lane & 15);
    int klo  = (lane >> 4) * 8;
    return load_frag_a_ptr(A + (size_t)m * lda + k0 + klo);
}

// B matrix 32(K)x16(N) from transposed weights Wt[n][k]. Lanes 0-15: N=lane, K 0..7/16..23; lanes 16-31: N=lane-16, K 8..15/24..31.
__device__ __forceinline__ v16bf load_frag_b(const bf16* Wt, int ldk, int n0, int k0) {
    int lane = threadIdx.x & 31;
    int n    = n0 + (lane & 15);
    int klo  = (lane >> 4) * 8;
    const bf16* base = Wt + (size_t)n * ldk + k0 + klo;
    v8bf lo = *(const v8bf*)(base);
    v8bf hi = *(const v8bf*)(base + 16);
    return __builtin_shufflevector(lo, hi, 0,1,2,3,4,5,6,7,8,9,10,11,12,13,14,15);
}

#define BROW 40   // LDS B-tile row stride in elements (80 B, bank-spread)

// B fragment from an LDS tile with row stride BROW elements.
__device__ __forceinline__ v16bf load_frag_b_lds(const bf16* tile, int n0) {
    int lane = threadIdx.x & 31;
    int n    = n0 + (lane & 15);
    int klo  = (lane >> 4) * 8;
    const bf16* base = tile + n * BROW + klo;
    v8bf lo = *(const v8bf*)(base);
    v8bf hi = *(const v8bf*)(base + 16);
    return __builtin_shufflevector(lo, hi, 0,1,2,3,4,5,6,7,8,9,10,11,12,13,14,15);
}

// ---------------- conversion kernels ----------------
__global__ __launch_bounds__(256) void convert_wt_kernel(
        const float* __restrict__ W, bf16* __restrict__ Wt, int K, int N, int KP, int NP) {
    size_t idx = (size_t)blockIdx.x * blockDim.x + threadIdx.x;
    size_t tot = (size_t)NP * KP;
    if (idx >= tot) return;
    int n = (int)(idx / KP);
    int k = (int)(idx % KP);
    float val = (n < N && k < K) ? W[(size_t)k * N + n] : 0.0f;
    Wt[idx] = (bf16)val;
}

__global__ __launch_bounds__(256) void pack_z_kernel(
        const float* __restrict__ x, bf16* __restrict__ zb) {
    int idx = blockIdx.x * blockDim.x + threadIdx.x;   // BATCH*LAT
    if (idx >= BATCH * LAT) return;
    int b = idx >> 7, k = idx & 127;
    zb[idx] = (bf16)x[(size_t)b * XW + (IN_F + 1) + k];
}

// ---------------- layer-1 GEMM (small): direct-from-global WMMA ----------------
__global__ __launch_bounds__(128, 1) void gemm1_kernel(
        const bf16* __restrict__ A, const bf16* __restrict__ Wt,
        const float* __restrict__ bias, bf16* __restrict__ out) {
    int wave = threadIdx.x >> 5;
    int lane = threadIdx.x & 31;
    int n0 = blockIdx.x * 64;
    int m0 = (blockIdx.y * 4 + wave) * 16;

    v8f acc[4] = {};
    for (int k0 = 0; k0 < LAT; k0 += 32) {
#pragma unroll
        for (int t = 0; t < 4; ++t) {
            v16bf a = load_frag_a(A, LAT, m0, k0);
            v16bf b = load_frag_b(Wt, LAT, n0 + 16 * t, k0);
            acc[t] = __builtin_amdgcn_wmma_f32_16x16x32_bf16(
                false, a, false, b, (short)0, acc[t], false, false);
        }
    }
    int mrow = m0 + (lane >> 4) * 8;
#pragma unroll
    for (int t = 0; t < 4; ++t) {
        int n = n0 + 16 * t + (lane & 15);
        float bs = bias[n];
#pragma unroll
        for (int r = 0; r < 8; ++r) {
            float val = tanhf(acc[t][r] + bs);
            out[(size_t)(mrow + r) * HID + n] = (bf16)val;
        }
    }
}

// ---------------- layer-2 GEMM: LDS-staged B (async double buffer), tanh epilogue ----------------
// Block: 4 waves x 32 rows = 128 rows, shared 64-col N tile. B tile 64x32 bf16 in LDS.
__global__ __launch_bounds__(128, 1) void gemm2_staged_kernel(
        const bf16* __restrict__ A, const bf16* __restrict__ Wt,
        const float* __restrict__ bias, bf16* __restrict__ out) {
    __shared__ bf16 tileB[2][64 * BROW];          // 10.2 KB
    int tid  = threadIdx.x;
    int wave = tid >> 5;
    int lane = tid & 31;
    int n0   = blockIdx.x * 64;
    int bm0  = (blockIdx.y * 4 + wave) * 32;

    // staging: 64 rows x 64 B = 256 x 16B chunks, 2 per thread
    const bf16* gsrc[2];
    bf16*       ldst[2][2];
#pragma unroll
    for (int j = 0; j < 2; ++j) {
        int c  = tid + 128 * j;
        int n  = c >> 2;
        int ko = (c & 3) * 8;
        gsrc[j]    = Wt + (size_t)(n0 + n) * HID + ko;
        ldst[j][0] = &tileB[0][n * BROW + ko];
        ldst[j][1] = &tileB[1][n * BROW + ko];
    }
    const bf16* abase0 = A + (size_t)(bm0 + (lane & 15)) * HID + (lane >> 4) * 8;
    const bf16* abase1 = abase0 + (size_t)16 * HID;

    v8f acc[2][4] = {};
    const int NSTEP = HID / 32;                   // 32 (even)

#pragma unroll
    for (int j = 0; j < 2; ++j) async_copy16(gsrc[j], ldst[j][0]);

    auto do_step = [&](int step, int buf) {
        async_wait0();
        __syncthreads();
        if (step + 1 < NSTEP) {
            int koff = (step + 1) * 32;
#pragma unroll
            for (int j = 0; j < 2; ++j) async_copy16(gsrc[j] + koff, ldst[j][buf ^ 1]);
        }
        v16bf a0 = load_frag_a_ptr(abase0 + step * 32);
        v16bf a1 = load_frag_a_ptr(abase1 + step * 32);
#pragma unroll
        for (int t = 0; t < 4; ++t) {
            v16bf bfrag = load_frag_b_lds(tileB[buf], 16 * t);
            acc[0][t] = __builtin_amdgcn_wmma_f32_16x16x32_bf16(
                false, a0, false, bfrag, (short)0, acc[0][t], false, false);
            acc[1][t] = __builtin_amdgcn_wmma_f32_16x16x32_bf16(
                false, a1, false, bfrag, (short)0, acc[1][t], false, false);
        }
    };
    for (int step = 0; step < NSTEP; step += 2) {
        do_step(step, 0);
        do_step(step + 1, 1);
    }

#pragma unroll
    for (int g = 0; g < 2; ++g) {
        int mrow = bm0 + g * 16 + (lane >> 4) * 8;
#pragma unroll
        for (int t = 0; t < 4; ++t) {
            int n = n0 + 16 * t + (lane & 15);
            float bs = (n < CS) ? bias[n] : 0.0f;
#pragma unroll
            for (int r = 0; r < 8; ++r) {
                float val = tanhf(acc[g][t][r] + bs);
                out[(size_t)(mrow + r) * CSP + n] = (bf16)val;
            }
        }
    }
}

// ---------------- compute v (B x 6) from the last 12 coeff columns ----------------
__global__ __launch_bounds__(256) void compute_v_kernel(
        const bf16* __restrict__ h2, const float* __restrict__ W3,
        const float* __restrict__ b3, const float* __restrict__ x,
        float* __restrict__ v) {
    __shared__ float red[256];
    __shared__ float cols[12];
    int b = blockIdx.x;
    int t = threadIdx.x;
    float acc[12];
#pragma unroll
    for (int c = 0; c < 12; ++c) acc[c] = 0.0f;
    for (int j = t; j < CS; j += 256) {
        float hv = (float)h2[(size_t)b * CSP + j];
        const float* wrow = W3 + (size_t)j * CS + (MM * KK);
#pragma unroll
        for (int c = 0; c < 12; ++c) acc[c] += hv * wrow[c];
    }
    for (int c = 0; c < 12; ++c) {
        red[t] = acc[c];
        __syncthreads();
        for (int s = 128; s > 0; s >>= 1) {
            if (t < s) red[t] += red[t + s];
            __syncthreads();
        }
        if (t == 0) cols[c] = red[0] + b3[MM * KK + c];
        __syncthreads();
    }
    if (t < 6) {
        float s  = x[(size_t)(BATCH - 1) * XW + IN_F];
        float sv = s * cols[t] + cols[6 + t];
        float nk = (float)(t % 3);
        v[b * KK + t] = (t < 3) ? cosf(sv * nk) : sinf(sv * nk);
    }
}

// ---------------- fused GEMM3 + v contraction ----------------
// Block: 4 waves x 32 batch rows = 128 rows, shared m-tile of 16 (96 coeff cols).
// B tile (96 x 32 bf16) staged in LDS via async loads, double buffered on ASYNCcnt.
#define G3_ROWS  32
#define G3_WAVES 4

__global__ __launch_bounds__(128, 1) void gemm3_fused_kernel(
        const bf16* __restrict__ h2, const bf16* __restrict__ Wt3,
        const float* __restrict__ b3, const float* __restrict__ v,
        float* __restrict__ w) {
    __shared__ bf16  tileB[2][96 * BROW];         // 15.4 KB
    __shared__ float ldsC[G3_WAVES][G3_ROWS][97]; // 49.7 KB
    int tid  = threadIdx.x;
    int wave = tid >> 5;
    int lane = tid & 31;
    int m0   = blockIdx.x * 16;                   // m-tile
    int c0   = m0 * KK;                           // coeff column base (96 cols)
    int bm0  = (blockIdx.y * G3_WAVES + wave) * G3_ROWS;

    // staging: 96 rows x 64 B = 384 x 16B chunks, 3 per thread
    const bf16* gsrc[3];
    bf16*       ldst[3][2];
#pragma unroll
    for (int j = 0; j < 3; ++j) {
        int c  = tid + 128 * j;
        int n  = c >> 2;
        int ko = (c & 3) * 8;
        gsrc[j]    = Wt3 + (size_t)(c0 + n) * CSP + ko;
        ldst[j][0] = &tileB[0][n * BROW + ko];
        ldst[j][1] = &tileB[1][n * BROW + ko];
    }
    const bf16* abase0 = h2 + (size_t)(bm0 + (lane & 15)) * CSP + (lane >> 4) * 8;
    const bf16* abase1 = abase0 + (size_t)16 * CSP;

    v8f acc[2][6] = {};
    const int NSTEP = CSP / 32;                   // 206 (even)

#pragma unroll
    for (int j = 0; j < 3; ++j) async_copy16(gsrc[j], ldst[j][0]);

    auto do_step = [&](int step, int buf) {
        async_wait0();                            // my async loads for buf done
        __syncthreads();                          // whole tile visible, prev reads done
        if (step + 1 < NSTEP) {
            int koff = (step + 1) * 32;
#pragma unroll
            for (int j = 0; j < 3; ++j) async_copy16(gsrc[j] + koff, ldst[j][buf ^ 1]);
        }
        v16bf a0 = load_frag_a_ptr(abase0 + step * 32);
        v16bf a1 = load_frag_a_ptr(abase1 + step * 32);
#pragma unroll
        for (int t = 0; t < 6; ++t) {
            v16bf bfrag = load_frag_b_lds(tileB[buf], 16 * t);
            acc[0][t] = __builtin_amdgcn_wmma_f32_16x16x32_bf16(
                false, a0, false, bfrag, (short)0, acc[0][t], false, false);
            acc[1][t] = __builtin_amdgcn_wmma_f32_16x16x32_bf16(
                false, a1, false, bfrag, (short)0, acc[1][t], false, false);
        }
    };
    for (int step = 0; step < NSTEP; step += 2) {
        do_step(step, 0);
        do_step(step + 1, 1);
    }

    int row = (lane >> 4) * 8;
    int col = lane & 15;
#pragma unroll
    for (int g = 0; g < 2; ++g)
#pragma unroll
        for (int t = 0; t < 6; ++t)
#pragma unroll
            for (int r = 0; r < 8; ++r)
                ldsC[wave][g * 16 + row + r][16 * t + col] = acc[g][t][r];
    __syncthreads();

    for (int idx = lane; idx < G3_ROWS * 16; idx += 32) {
        int br = idx >> 4, m = idx & 15;
        int bg = bm0 + br;
        int mg = m0 + m;
        float sum = 0.0f;
#pragma unroll
        for (int k = 0; k < 6; ++k) {
            float vk = v[bg * KK + k];
            sum += (ldsC[wave][br][m * KK + k] + b3[mg * KK + k]) * vk;
        }
        w[(size_t)bg * MM + mg] = sum;
    }
}

// ---------------- final: out[b,o] = sum_i x[b,i]*w[b,i*64+o] + w[b,1024+o] ----------------
__global__ __launch_bounds__(256) void final_kernel(
        const float* __restrict__ x, const float* __restrict__ w,
        float* __restrict__ out) {
    int idx = blockIdx.x * blockDim.x + threadIdx.x;   // BATCH*OUT_F
    if (idx >= BATCH * OUT_F) return;
    int b = idx >> 6, o = idx & 63;
    const float* wr = w + (size_t)b * MM;
    float sum = wr[IN_F * OUT_F + o];
#pragma unroll
    for (int i = 0; i < IN_F; ++i)
        sum += x[(size_t)b * XW + i] * wr[i * OUT_F + o];
    out[idx] = sum;
}

// ---------------- host launcher ----------------
extern "C" void kernel_launch(void* const* d_in, const int* in_sizes, int n_in,
                              void* d_out, int out_size, void* d_ws, size_t ws_size,
                              hipStream_t stream) {
    const float* x  = (const float*)d_in[0];
    const float* W1 = (const float*)d_in[1];
    const float* b1 = (const float*)d_in[2];
    const float* W2 = (const float*)d_in[3];
    const float* b2 = (const float*)d_in[4];
    const float* W3 = (const float*)d_in[5];
    const float* b3 = (const float*)d_in[6];
    float* out = (float*)d_out;

    char* ws = (char*)d_ws;
    size_t off = 0;
    auto take = [&](size_t bytes) -> void* {
        void* p = ws + off;
        off += (bytes + 255) & ~(size_t)255;
        return p;
    };
    bf16* Wt1 = (bf16*)take((size_t)HID * LAT * 2);
    bf16* Wt2 = (bf16*)take((size_t)CSP * HID * 2);
    bf16* Wt3 = (bf16*)take((size_t)CSP * CSP * 2);
    bf16* zb  = (bf16*)take((size_t)BATCH * LAT * 2);
    bf16* h1  = (bf16*)take((size_t)BATCH * HID * 2);
    bf16* h2  = (bf16*)take((size_t)BATCH * CSP * 2);
    float* vv = (float*)take((size_t)BATCH * KK * 4);
    float* w  = (float*)take((size_t)BATCH * MM * 4);

    {
        size_t tot1 = (size_t)HID * LAT;
        convert_wt_kernel<<<(unsigned)((tot1 + 255) / 256), 256, 0, stream>>>(W1, Wt1, LAT, HID, LAT, HID);
        size_t tot2 = (size_t)CSP * HID;
        convert_wt_kernel<<<(unsigned)((tot2 + 255) / 256), 256, 0, stream>>>(W2, Wt2, HID, CS, HID, CSP);
        size_t tot3 = (size_t)CSP * CSP;
        convert_wt_kernel<<<(unsigned)((tot3 + 255) / 256), 256, 0, stream>>>(W3, Wt3, CS, CS, CSP, CSP);
    }
    pack_z_kernel<<<(BATCH * LAT + 255) / 256, 256, 0, stream>>>(x, zb);

    // h1 = tanh(z @ W1 + b1)
    gemm1_kernel<<<dim3(HID / 64, BATCH / 64), 128, 0, stream>>>(zb, Wt1, b1, h1);

    // h2 = tanh(h1 @ W2 + b2), LDS-staged B
    gemm2_staged_kernel<<<dim3(CSP / 64, BATCH / 128), 128, 0, stream>>>(h1, Wt2, b2, h2);

    compute_v_kernel<<<BATCH, 256, 0, stream>>>(h2, W3, b3, x, vv);

    gemm3_fused_kernel<<<dim3(MM / 16, BATCH / (G3_ROWS * G3_WAVES)), 128, 0, stream>>>(
        h2, Wt3, b3, vv, w);

    final_kernel<<<(BATCH * OUT_F + 255) / 256, 256, 0, stream>>>(x, w, out);
}
